// DecisionMaking_83906481094722
// MI455X (gfx1250) — matrix operand order repaired and checked by reading, not compiled, not run on hardware
//
#include <hip/hip_runtime.h>
#include <hip/hip_bf16.h>
#include <stdint.h>

// ---------------------------------------------------------------------------
// CDNA5 / gfx1250 implementation (wave32, WMMA).
// - All GEMMs: v_wmma_f32_16x16x32_f16. W panel staged in LDS shared by 8
//   waves/block. All 8 B fragments for a k-step are loaded into distinct
//   registers with a sched_barrier before the 8-WMMA burst: kills the
//   WMMA->VALU WAR hazard NOPs and the per-WMMA dscnt waits.
// - Actor MLP fused: layer-0 decomposed into broadcast terms (P_o/P_m/P_p);
//   tanh -> 128x128 WMMA layer-1 (A1 resident in LDS) -> tanh -> A2 dot,
//   never materializing the 134MB [B,NM,NO,128] activation tensor.
// ---------------------------------------------------------------------------

typedef _Float16 f16;
typedef __attribute__((ext_vector_type(16))) _Float16 v16h;
typedef __attribute__((ext_vector_type(8)))  float    v8f;
typedef __attribute__((ext_vector_type(4)))  float    vf4;

union Frag { v16h v; vf4 q[2]; };

static constexpr int Bb   = 16;
static constexpr int NO   = 512;
static constexpr int NM   = 32;
static constexpr int DO   = 128;
static constexpr int DM   = 64;
static constexpr int Hh   = 64;
static constexpr int Kk   = 16;
static constexpr int C8   = 1024;   // DO*8
static constexpr int HID  = 128;

// ---- WMMA fragment loaders (per CDNA5 ISA 7.12.2 layouts) -----------------
// A 16x32 f16: lane&15 = M; lanes<16 hold K {0..7,16..23}, lanes>=16 {8..15,24..31}
__device__ inline v16h load_a_frag(const f16* X, int ld, int m0, int k0, int lane) {
  int half = lane >> 4;
  const f16* p = X + (size_t)(m0 + (lane & 15)) * ld + k0 + half * 8;
  Frag u;
  u.q[0] = *(const vf4*)(p);        // K = half*8 + 0..7
  u.q[1] = *(const vf4*)(p + 16);   // K = 16 + half*8 + 0..7
  return u.v;
}
// B 32x16 f16 from W[N,K] row-major (B[k,n] = W[n,k]):
// lane&15 = N; lanes<16 hold K {0..7,16..23}, lanes>=16 {8..15,24..31}
__device__ inline v16h load_b_frag(const f16* W, int ld, int n0, int k0, int lane) {
  int half = lane >> 4;
  const f16* p = W + (size_t)(n0 + (lane & 15)) * ld + k0 + half * 8;
  Frag u;
  u.q[0] = *(const vf4*)(p);
  u.q[1] = *(const vf4*)(p + 16);
  return u.v;
}

// ---- Generic Y = X @ W^T (+bias), 16 rows x 128 cols per wave -------------
// X: [M,K] f16 row-major (ldx), W: [N,K] f16 row-major (ldw, torch Linear).
// W panel (128 cols x <=128 K) staged in LDS, shared by 8 waves.
// grid = (M/128, N/128), block = 256.  M must be a multiple of 128.
__global__ __launch_bounds__(256) void gemm16x128(
    const f16* __restrict__ X, int ldx,
    const f16* __restrict__ W, int ldw,
    const float* __restrict__ bias,
    float* __restrict__ Yf, f16* __restrict__ Yh, int ldy,
    int Kdim)
{
  __shared__ alignas(16) f16 wtile[128 * 128];        // 32 KB
  int lane = threadIdx.x & 31, wave = threadIdx.x >> 5;
  int m0 = (blockIdx.x * 8 + wave) * 16;
  int n0 = blockIdx.y * 128;

  v8f acc[8];
  v8f z = {0.f,0.f,0.f,0.f,0.f,0.f,0.f,0.f};
#pragma unroll
  for (int i = 0; i < 8; ++i) acc[i] = z;

  for (int kc = 0; kc < Kdim; kc += 128) {
    int KC = (Kdim - kc) < 128 ? (Kdim - kc) : 128;
    int cpr = KC >> 3;                               // 8-half chunks per row
    __syncthreads();                                 // previous panel consumed
    for (int idx = threadIdx.x; idx < 128 * cpr; idx += 256) {
      int n = idx / cpr, k8 = (idx % cpr) * 8;
      *(vf4*)&wtile[n * KC + k8] =
          *(const vf4*)(W + (size_t)(n0 + n) * ldw + kc + k8);
    }
    __syncthreads();
    for (int k0 = 0; k0 < KC; k0 += 32) {
      // issue A (global) + all 8 B fragments (LDS) first, in distinct regs
      v16h a = load_a_frag(X, ldx, m0, kc + k0, lane);
      v16h bf[8];
#pragma unroll
      for (int i = 0; i < 8; ++i) bf[i] = load_b_frag(wtile, KC, 16 * i, k0, lane);
      __builtin_amdgcn_sched_barrier(0);             // keep loads ahead of WMMAs
#pragma unroll
      for (int i = 0; i < 8; ++i)
        acc[i] = __builtin_amdgcn_wmma_f32_16x16x32_f16(
            false, a, false, bf[i], (short)0, acc[i], false, false);
    }
  }
  int half = lane >> 4, nl = lane & 15;
#pragma unroll
  for (int i = 0; i < 8; ++i) {
    int n = n0 + 16 * i + nl;
    float bv = bias ? bias[n] : 0.0f;
#pragma unroll
    for (int r = 0; r < 8; ++r) {
      float v = acc[i][r] + bv;
      size_t idx = (size_t)(m0 + r + 8 * half) * ldy + n;
      if (Yf) Yf[idx] = v;
      if (Yh) Yh[idx] = (f16)v;
    }
  }
}

// ---- elementwise f32 -> f16 ----------------------------------------------
__global__ void cvt_f32_f16(const float* __restrict__ s, f16* __restrict__ d, int n) {
  int i = blockIdx.x * blockDim.x + threadIdx.x;
  if (i < n) d[i] = (f16)s[i];
}

// ---- external attention: per-head K=16 linear (l0) ------------------------
__global__ void ea_l0(const f16* __restrict__ T, const float* __restrict__ Wl0,
                      const float* __restrict__ bl0, float* __restrict__ attn) {
  int t = blockIdx.x * blockDim.x + threadIdx.x;        // B*H*NO = 524288
  if (t >= Bb * Hh * NO) return;
  int n = t & 511, h = (t >> 9) & 63, b = t >> 15;
  const f16* x = T + ((size_t)(b * NO + n)) * C8 + h * Kk;
  float xv[16];
#pragma unroll
  for (int j = 0; j < 16; ++j) xv[j] = (float)x[j];
  float* o = attn + ((size_t)((b * Hh + h) * NO + n)) * Kk;
#pragma unroll
  for (int k = 0; k < 16; ++k) {
    float s = bl0[k];
#pragma unroll
    for (int j = 0; j < 16; ++j) s += xv[j] * Wl0[k * 16 + j];
    o[k] = s;
  }
}

// ---- softmax over N (axis=-2) of attn[B,H,N,K]; one block per (b,h,k) ----
__global__ __launch_bounds__(128) void ea_softmax_n(float* __restrict__ attn) {
  int c = blockIdx.x;                    // (b*H+h)*16 + k, 16384 columns
  int bh = c >> 4, k = c & 15;
  float* p = attn + (size_t)bh * NO * Kk + k;
  __shared__ float red[128];
  int tid = threadIdx.x;
  float mx = -3.0e38f;
  for (int n = tid; n < NO; n += 128) mx = fmaxf(mx, p[n * Kk]);
  red[tid] = mx; __syncthreads();
  for (int s = 64; s > 0; s >>= 1) { if (tid < s) red[tid] = fmaxf(red[tid], red[tid + s]); __syncthreads(); }
  float MX = red[0]; __syncthreads();
  float sm = 0.f;
  for (int n = tid; n < NO; n += 128) sm += expf(p[n * Kk] - MX);
  red[tid] = sm; __syncthreads();
  for (int s = 64; s > 0; s >>= 1) { if (tid < s) red[tid] += red[tid + s]; __syncthreads(); }
  float inv = 1.0f / red[0];
  for (int n = tid; n < NO; n += 128) p[n * Kk] = expf(p[n * Kk] - MX) * inv;
}

// ---- attn/(1e-10+sum_k) , then l1 linear, transpose to [B,N,H*K] f16 -----
__global__ void ea_norm_l1(const float* __restrict__ attn, const float* __restrict__ Wl1,
                           const float* __restrict__ bl1, f16* __restrict__ xl1) {
  int t = blockIdx.x * blockDim.x + threadIdx.x;
  if (t >= Bb * Hh * NO) return;
  int n = t & 511, h = (t >> 9) & 63, b = t >> 15;
  const float* a = attn + ((size_t)((b * Hh + h) * NO + n)) * Kk;
  float av[16]; float s = 0.f;
#pragma unroll
  for (int j = 0; j < 16; ++j) { av[j] = a[j]; s += av[j]; }
  float inv = 1.0f / (1e-10f + s);
#pragma unroll
  for (int j = 0; j < 16; ++j) av[j] *= inv;
  f16* o = xl1 + ((size_t)(b * NO + n)) * C8 + h * Kk;
#pragma unroll
  for (int k = 0; k < 16; ++k) {
    float sv = bl1[k];
#pragma unroll
    for (int j = 0; j < 16; ++j) sv += av[j] * Wl1[k * 16 + j];
    o[k] = (f16)sv;
  }
}

// ---- GAT: row dot with alpha vector (a_o, a_m) ----------------------------
__global__ void rowdot128(const float* __restrict__ src, const float* __restrict__ alpha,
                          float* __restrict__ out, int R) {
  int r = blockIdx.x * blockDim.x + threadIdx.x;
  if (r >= R) return;
  const float* p = src + (size_t)r * 128;
  float s = 0.f;
  for (int c = 0; c < 128; ++c) s += p[c] * alpha[c];
  out[r] = s;
}

// ---- GAT aggregation: masked softmax over opes + einsum + residual -------
__global__ __launch_bounds__(128) void gat_agg(
    const float* __restrict__ adj, const float* __restrict__ a_o, const float* __restrict__ a_m,
    const float* __restrict__ hog, const float* __restrict__ hmac,
    float* __restrict__ hmacs, f16* __restrict__ hmacs16)
{
  int bm = blockIdx.x; int b = bm >> 5, m = bm & 31;
  __shared__ float alph[NO];
  __shared__ float red[128];
  int tid = threadIdx.x;
  float am = a_m[b * NM + m];
  float mx = -3.0e38f;
  for (int o = tid; o < NO; o += 128) {
    float ad = adj[((size_t)(b * NO + o)) * NM + m];
    float ev = ad * (a_o[b * NO + o] + am);
    ev = ev >= 0.f ? ev : 0.2f * ev;                 // leaky_relu(0.2)
    ev = (ad == 1.0f) ? ev : -9e10f;                 // mask
    alph[o] = ev; mx = fmaxf(mx, ev);
  }
  red[tid] = mx; __syncthreads();
  for (int s = 64; s > 0; s >>= 1) { if (tid < s) red[tid] = fmaxf(red[tid], red[tid + s]); __syncthreads(); }
  float MX = red[0]; __syncthreads();
  float sm = 0.f;
  for (int o = tid; o < NO; o += 128) {
    float ex = expf(alph[o] - MX);
    sm += ex;
    float ad = adj[((size_t)(b * NO + o)) * NM + m];
    alph[o] = (ad == 1.0f) ? ex : 0.f;               // softmax * mask (numerator)
  }
  red[tid] = sm; __syncthreads();
  for (int s = 64; s > 0; s >>= 1) { if (tid < s) red[tid] += red[tid + s]; __syncthreads(); }
  float inv = 1.0f / red[0];
  __syncthreads();
  int d = tid;                                        // 128 features
  float acc = 0.f;
  for (int o = 0; o < NO; ++o) acc += alph[o] * hog[((size_t)(b * NO + o)) * 128 + d];
  size_t idx = ((size_t)(b * NM + m)) * 128 + d;
  float v = acc * inv + hmac[idx];
  hmacs[idx] = v; hmacs16[idx] = (f16)v;
}

// ---- mean pooling over rows ----------------------------------------------
__global__ void pool_mean(const float* __restrict__ src, float* __restrict__ out, int R) {
  int b = blockIdx.x, d = threadIdx.x;                // 128 threads
  float s = 0.f;
  for (int r = 0; r < R; ++r) s += src[((size_t)(b * R + r)) * 128 + d];
  out[b * 128 + d] = s / (float)R;
}

// ---- P_p[b,h] = pooled terms of decomposed actor layer-0 + Ab0 -----------
__global__ void pp_kernel(const float* __restrict__ hop_p, const float* __restrict__ hmac_p,
                          const float* __restrict__ A0, const float* __restrict__ Ab0,
                          float* __restrict__ Pp) {
  int b = blockIdx.x, h = threadIdx.x;                // 128 threads
  float s = Ab0[h];
  const float* w = A0 + (size_t)h * 512;
  for (int c = 0; c < 128; ++c)
    s += hop_p[b * 128 + c] * w[256 + c] + hmac_p[b * 128 + c] * w[384 + c];
  Pp[b * 128 + h] = s;
}

// ---- fused actor: tanh(P_o+P_m+P_p) -> WMMA 128x128 -> tanh -> A2 dot ----
// A1 (32KB) resident in LDS per block; each of 8 waves owns a 16-row tile.
// grid = 2048 blocks x 256 threads.
__global__ __launch_bounds__(256) void actor_kernel(
    const float* __restrict__ Po, const float* __restrict__ Pm, const float* __restrict__ Pp,
    const f16* __restrict__ A1h, const float* __restrict__ Ab1,
    const float* __restrict__ A2, const float* __restrict__ Ab2,
    float* __restrict__ logits)
{
  __shared__ alignas(16) f16 a1t[128 * 128];          // 32 KB, shared weights
  __shared__ alignas(16) f16 zb[8][16 * 128];         // 32 KB, per-wave tiles
  int lane = threadIdx.x & 31, wave = threadIdx.x >> 5;
  int tile = blockIdx.x * 8 + wave;                    // 0..16383
  int rowBase = tile * 16;
  int b   = rowBase >> 14;                             // / (NM*NO)
  int rem = rowBase & 16383;
  int nm  = rem >> 9;
  int no0 = rem & 511;

  // stage A1 once per block (coalesced b128 loads)
  for (int idx = threadIdx.x; idx < 2048; idx += 256) {
    int n = idx >> 4, k8 = (idx & 15) * 8;
    *(vf4*)&a1t[n * 128 + k8] = *(const vf4*)(A1h + n * 128 + k8);
  }
  __builtin_prefetch(Po + ((size_t)(b * NO + no0)) * 128, 0, 3); // global_prefetch_b8

  f16* zt = zb[wave];
  const float* po = Po + ((size_t)(b * NO + no0)) * 128;
  const float* pm = Pm + ((size_t)(b * NM + nm)) * 128;
  const float* pp = Pp + b * 128;
  for (int t = lane; t < 16 * 128; t += 32) {
    int m = t >> 7, c = t & 127;
    float v = po[m * 128 + c] + pm[c] + pp[c];
    zt[t] = (f16)tanhf(v);
  }
  __syncthreads();

  v8f acc[8];
  v8f zz = {0.f,0.f,0.f,0.f,0.f,0.f,0.f,0.f};
#pragma unroll
  for (int i = 0; i < 8; ++i) acc[i] = zz;
#pragma unroll
  for (int k0 = 0; k0 < 128; k0 += 32) {
    v16h a = load_a_frag(zt, 128, 0, k0, lane);        // LDS
    v16h bf[8];
#pragma unroll
    for (int i = 0; i < 8; ++i) bf[i] = load_b_frag(a1t, 128, 16 * i, k0, lane);
    __builtin_amdgcn_sched_barrier(0);                 // loads ahead of WMMAs
#pragma unroll
    for (int i = 0; i < 8; ++i)
      acc[i] = __builtin_amdgcn_wmma_f32_16x16x32_f16(
          false, a, false, bf[i], (short)0, acc[i], false, false);
  }
  int half = lane >> 4, nl = lane & 15;
  float ab1v[8], a2v[8];
#pragma unroll
  for (int i = 0; i < 8; ++i) { ab1v[i] = Ab1[16 * i + nl]; a2v[i] = A2[16 * i + nl]; }
  float ab2 = Ab2[0];
#pragma unroll
  for (int r = 0; r < 8; ++r) {
    float p = 0.f;
#pragma unroll
    for (int i = 0; i < 8; ++i) p += tanhf(acc[i][r] + ab1v[i]) * a2v[i];
    p += __shfl_xor(p, 1);
    p += __shfl_xor(p, 2);
    p += __shfl_xor(p, 4);
    p += __shfl_xor(p, 8);                             // sum over 16-lane half
    if (nl == 0) {
      int no = no0 + r + 8 * half;
      logits[(size_t)b * (NM * NO) + nm * NO + no] = p + ab2;
    }
  }
}

// ---- masked log-softmax over 16384 actions, entropy, gather --------------
__global__ __launch_bounds__(256) void final_kernel(
    const float* __restrict__ logits, const unsigned char* __restrict__ mask,
    const int* __restrict__ aidx, float* __restrict__ out)
{
  int b = blockIdx.x, tid = threadIdx.x;
  __shared__ float red[256];
  const int NA = NM * NO;
  float mx = -3.0e38f;
  for (int j = tid; j < NA; j += 256) {
    int nm = j >> 9, no = j & 511;
    bool mk = mask[((size_t)(b * NO + no)) * NM + nm] != 0;
    float lg = mk ? logits[(size_t)b * NA + j] : -1e30f;
    mx = fmaxf(mx, lg);
  }
  red[tid] = mx; __syncthreads();
  for (int s = 128; s > 0; s >>= 1) { if (tid < s) red[tid] = fmaxf(red[tid], red[tid + s]); __syncthreads(); }
  float MX = red[0]; __syncthreads();
  float sm = 0.f;
  for (int j = tid; j < NA; j += 256) {
    int nm = j >> 9, no = j & 511;
    bool mk = mask[((size_t)(b * NO + no)) * NM + nm] != 0;
    float lg = mk ? logits[(size_t)b * NA + j] : -1e30f;
    sm += expf(lg - MX);
  }
  red[tid] = sm; __syncthreads();
  for (int s = 128; s > 0; s >>= 1) { if (tid < s) red[tid] += red[tid + s]; __syncthreads(); }
  float logZ = MX + logf(red[0]); __syncthreads();
  float ent = 0.f;
  for (int j = tid; j < NA; j += 256) {
    int nm = j >> 9, no = j & 511;
    bool mk = mask[((size_t)(b * NO + no)) * NM + nm] != 0;
    if (mk) { float lp = logits[(size_t)b * NA + j] - logZ; ent += expf(lp) * lp; }
  }
  red[tid] = ent; __syncthreads();
  for (int s = 128; s > 0; s >>= 1) { if (tid < s) red[tid] += red[tid + s]; __syncthreads(); }
  if (tid == 0) {
    int j = aidx[b];
    int nm = j >> 9, no = j & 511;
    bool mk = mask[((size_t)(b * NO + no)) * NM + nm] != 0;
    float lg = mk ? logits[(size_t)b * NA + j] : -1e30f;
    out[b]      = lg - logZ;       // action_logprobs
    out[32 + b] = -red[0];         // dist_entropys
  }
}

// ---- critic MLP 256 -> 128 -> 128 -> 1 ------------------------------------
__global__ __launch_bounds__(128) void critic_kernel(
    const float* __restrict__ hop_p, const float* __restrict__ hmac_p,
    const float* __restrict__ C0, const float* __restrict__ Cb0,
    const float* __restrict__ C1, const float* __restrict__ Cb1,
    const float* __restrict__ C2, const float* __restrict__ Cb2,
    float* __restrict__ out)
{
  int b = blockIdx.x, tid = threadIdx.x;
  __shared__ float x[256]; __shared__ float t1[128]; __shared__ float t2[128]; __shared__ float red[128];
  x[tid] = hop_p[b * 128 + tid]; x[128 + tid] = hmac_p[b * 128 + tid];
  __syncthreads();
  float s = Cb0[tid];
  for (int c = 0; c < 256; ++c) s += C0[tid * 256 + c] * x[c];
  t1[tid] = tanhf(s); __syncthreads();
  s = Cb1[tid];
  for (int c = 0; c < 128; ++c) s += C1[tid * 128 + c] * t1[c];
  t2[tid] = tanhf(s); __syncthreads();
  red[tid] = C2[tid] * t2[tid]; __syncthreads();
  for (int st = 64; st > 0; st >>= 1) { if (tid < st) red[tid] += red[tid + st]; __syncthreads(); }
  if (tid == 0) out[16 + b] = red[0] + Cb2[0];         // state_values
}

// ===========================================================================
extern "C" void kernel_launch(void* const* d_in, const int* in_sizes, int n_in,
                              void* d_out, int out_size, void* d_ws, size_t ws_size,
                              hipStream_t stream) {
  (void)in_sizes; (void)n_in; (void)out_size; (void)ws_size;
  const float* adj     = (const float*)d_in[0];
  const float* opes    = (const float*)d_in[1];
  const float* macs    = (const float*)d_in[2];
  const unsigned char* maskp = (const unsigned char*)d_in[3];
  const int*   aidx    = (const int*)d_in[4];
  const float* Wtrans  = (const float*)d_in[5];
  const float* btrans  = (const float*)d_in[6];
  const float* Wl0     = (const float*)d_in[7];
  const float* bl0     = (const float*)d_in[8];
  const float* Wl1     = (const float*)d_in[9];
  const float* bl1     = (const float*)d_in[10];
  const float* Wproj   = (const float*)d_in[11];
  const float* bproj   = (const float*)d_in[12];
  const float* Wo      = (const float*)d_in[13];
  const float* Wm      = (const float*)d_in[14];
  const float* alpha_o = (const float*)d_in[15];
  const float* alpha_m = (const float*)d_in[16];
  const float* A0      = (const float*)d_in[17];
  const float* Ab0     = (const float*)d_in[18];
  const float* A1      = (const float*)d_in[19];
  const float* Ab1     = (const float*)d_in[20];
  const float* A2      = (const float*)d_in[21];
  const float* Ab2     = (const float*)d_in[22];
  const float* C0      = (const float*)d_in[23];
  const float* Cb0     = (const float*)d_in[24];
  const float* C1      = (const float*)d_in[25];
  const float* Cb1     = (const float*)d_in[26];
  const float* C2      = (const float*)d_in[27];
  const float* Cb2     = (const float*)d_in[28];
  float* out = (float*)d_out;

  char* ws = (char*)d_ws;
  size_t off = 0;
  auto alloc = [&](size_t bytes) -> char* {
    char* p = ws + off;
    off = (off + bytes + 255) & ~(size_t)255;
    return p;
  };
  const int MO = Bb * NO;   // 8192 ope rows
  const int MM = Bb * NM;   // 512 mac rows

  f16*   opes16   = (f16*)  alloc((size_t)MO * DO * 2);
  f16*   macs16   = (f16*)  alloc((size_t)MM * DM * 2);
  f16*   Wtrans16 = (f16*)  alloc((size_t)C8 * DO * 2);
  f16*   Wproj16  = (f16*)  alloc((size_t)DO * C8 * 2);
  f16*   Wo16     = (f16*)  alloc((size_t)DO * DO * 2);
  f16*   Wm16     = (f16*)  alloc((size_t)DO * DM * 2);
  f16*   A016     = (f16*)  alloc((size_t)HID * 512 * 2);
  f16*   A116     = (f16*)  alloc((size_t)HID * HID * 2);
  f16*   T16      = (f16*)  alloc((size_t)MO * C8 * 2);
  float* attn     = (float*)alloc((size_t)Bb * Hh * NO * Kk * 4);
  f16*   xl1      = (f16*)  alloc((size_t)MO * C8 * 2);
  float* hopes32  = (float*)alloc((size_t)MO * DO * 4);
  f16*   hopes16  = (f16*)  alloc((size_t)MO * DO * 2);
  float* hog32    = (float*)alloc((size_t)MO * DO * 4);
  float* hmac32   = (float*)alloc((size_t)MM * DO * 4);
  float* hmacs32  = (float*)alloc((size_t)MM * DO * 4);
  f16*   hmacs16  = (f16*)  alloc((size_t)MM * DO * 2);
  float* a_o      = (float*)alloc((size_t)MO * 4);
  float* a_m      = (float*)alloc((size_t)MM * 4);
  float* hop_pool = (float*)alloc((size_t)Bb * DO * 4);
  float* hmac_pool= (float*)alloc((size_t)Bb * DO * 4);
  float* Po       = (float*)alloc((size_t)MO * HID * 4);
  float* Pm       = (float*)alloc((size_t)MM * HID * 4);
  float* Pp       = (float*)alloc((size_t)Bb * HID * 4);
  float* logits   = (float*)alloc((size_t)Bb * NM * NO * 4);

  auto cvt = [&](const float* s, f16* d, int n) {
    cvt_f32_f16<<<(n + 255) / 256, 256, 0, stream>>>(s, d, n);
  };
  // f32 -> f16 staging of activations and GEMM weights
  cvt(opes,   opes16,   MO * DO);
  cvt(macs,   macs16,   MM * DM);
  cvt(Wtrans, Wtrans16, C8 * DO);
  cvt(Wproj,  Wproj16,  DO * C8);
  cvt(Wo,     Wo16,     DO * DO);
  cvt(Wm,     Wm16,     DO * DM);
  cvt(A0,     A016,     HID * 512);
  cvt(A1,     A116,     HID * HID);

  // external attention: trans GEMM [8192,128]x[1024,128]^T -> T16 (f16)
  gemm16x128<<<dim3(MO / 128, C8 / 128), 256, 0, stream>>>(
      opes16, DO, Wtrans16, DO, btrans, nullptr, T16, C8, DO);
  // per-head l0 (K=16), softmax over N, normalize + l1 + transpose
  ea_l0<<<(Bb * Hh * NO + 255) / 256, 256, 0, stream>>>(T16, Wl0, bl0, attn);
  ea_softmax_n<<<Bb * Hh * Kk, 128, 0, stream>>>(attn);
  ea_norm_l1<<<(Bb * Hh * NO + 255) / 256, 256, 0, stream>>>(attn, Wl1, bl1, xl1);
  // proj GEMM [8192,1024]x[128,1024]^T -> h_opes (f32 + f16)
  gemm16x128<<<dim3(MO / 128, 1), 256, 0, stream>>>(
      xl1, C8, Wproj16, C8, bproj, hopes32, hopes16, DO, C8);

  // GAT projections
  gemm16x128<<<dim3(MO / 128, 1), 256, 0, stream>>>(
      opes16, DO, Wo16, DO, nullptr, hog32, nullptr, DO, DO);
  gemm16x128<<<dim3(MM / 128, 1), 256, 0, stream>>>(
      macs16, DM, Wm16, DM, nullptr, hmac32, nullptr, DO, DM);
  rowdot128<<<(MO + 127) / 128, 128, 0, stream>>>(hog32, alpha_o, a_o, MO);
  rowdot128<<<(MM + 127) / 128, 128, 0, stream>>>(hmac32, alpha_m, a_m, MM);
  gat_agg<<<Bb * NM, 128, 0, stream>>>(adj, a_o, a_m, hog32, hmac32, hmacs32, hmacs16);

  // pooling
  pool_mean<<<Bb, 128, 0, stream>>>(hopes32, hop_pool, NO);
  pool_mean<<<Bb, 128, 0, stream>>>(hmacs32, hmac_pool, NM);

  // actor layer-0 decomposition
  gemm16x128<<<dim3(MO / 128, 1), 256, 0, stream>>>(
      hopes16, DO, A016 + 0, 512, nullptr, Po, nullptr, HID, DO);
  gemm16x128<<<dim3(MM / 128, 1), 256, 0, stream>>>(
      hmacs16, DO, A016 + 128, 512, nullptr, Pm, nullptr, HID, DO);
  pp_kernel<<<Bb, 128, 0, stream>>>(hop_pool, hmac_pool, A0, Ab0, Pp);

  // fused actor MLP over all 262144 actions (16384 tiles, 8 waves/block)
  actor_kernel<<<(Bb * NM * NO / 16) / 8, 256, 0, stream>>>(
      Po, Pm, Pp, A116, Ab1, A2, Ab2, logits);

  // masked log-softmax / entropy / gather + critic
  final_kernel<<<Bb, 256, 0, stream>>>(logits, maskp, aidx, out);
  critic_kernel<<<Bb, 128, 0, stream>>>(hop_pool, hmac_pool,
                                        C0, Cb0, C1, Cb1, C2, Cb2, out);
}
